// ARISEMultiHeadAttention_2929167696293
// MI455X (gfx1250) — matrix operand order repaired and checked by reading, not compiled, not run on hardware
//
#include <hip/hip_runtime.h>
#include <hip/hip_bf16.h>

#define D_MODEL  1024
#define N_HEADS  16
#define HEAD_DIM 64
#define BATCH    2
#define SEQ      2048

typedef __attribute__((ext_vector_type(16))) __bf16         v16bf;
typedef __attribute__((ext_vector_type(8)))  float          v8f;
typedef __attribute__((ext_vector_type(16))) unsigned short v16u;
typedef __attribute__((ext_vector_type(8)))  unsigned short v8u;

union FragBF {
    v8u   h[2];
    v16u  u;
    v16bf bf;
};

__device__ __forceinline__ unsigned short f32_to_bf16_rne(float f) {
    union { float f; unsigned u; } x;
    x.f = f;
    unsigned u = x.u;
    u += 0x7fffu + ((u >> 16) & 1u);   // round-to-nearest-even
    return (unsigned short)(u >> 16);
}

// A fragment: 16x32 bf16 (MxK). Lane L (L<16): row M=L, K = k..k+7 and k+16..k+23.
// Lane L (L>=16): row M=L-16, K = k+8..k+15 and k+24..k+31.
__device__ __forceinline__ v16bf load_a_bf16(const unsigned short* A, int lda,
                                             int m0, int k, int lane) {
    const unsigned short* p =
        A + (size_t)(m0 + (lane & 15)) * lda + k + ((lane >> 4) << 3);
    FragBF f;
    f.h[0] = *(const v8u*)(p);
    f.h[1] = *(const v8u*)(p + 16);
    return f.bf;
}

// Same A fragment but source is f32 (attn matrix); convert to bf16 in-register.
__device__ __forceinline__ v16bf load_a_f32(const float* A, int lda,
                                            int m0, int k, int lane) {
    typedef __attribute__((ext_vector_type(8))) float v8fv;
    const float* p = A + (size_t)(m0 + (lane & 15)) * lda + k + ((lane >> 4) << 3);
    v8fv lo = *(const v8fv*)(p);
    v8fv hi = *(const v8fv*)(p + 16);
    FragBF f;
#pragma unroll
    for (int i = 0; i < 8; ++i) {
        f.u[i]     = f32_to_bf16_rne(lo[i]);
        f.u[i + 8] = f32_to_bf16_rne(hi[i]);
    }
    return f.bf;
}

// B fragment: 32x16 bf16 (KxN), operand stored row-major [N,K] (y = x @ W^T).
// Lane L: column N = L&15, K = k + (L>=16 ? 8 : 0) .. +7 and +16 .. +23 (same as A).
__device__ __forceinline__ v16bf load_b_bf16(const unsigned short* B, int ldb,
                                             int n0, int k, int lane) {
    const unsigned short* p =
        B + (size_t)(n0 + (lane & 15)) * ldb + k + ((lane >> 4) << 3);
    FragBF f;
    f.h[0] = *(const v8u*)(p);
    f.h[1] = *(const v8u*)(p + 16);
    return f.bf;
}

// Generic batched GEMM: C[z] = scale * A[z] @ B[z]^T  (A:[M,K], B:[N,K], K-major).
// z = b*16 + h; per-operand element offsets are base + b*SB + h*SH.
// Wave tile 64x64 (4x4 WMMA frags); block = WROWS x WCOLS waves
// -> block tile (WROWS*64) x (WCOLS*64). ALL dimensions must divide exactly:
// no bounds checks, so WMMA runs with EXEC all-ones (ISA requirement).
template <int WROWS, int WCOLS, bool A_F32, bool OUT_BF16, bool TRANS_OUT>
__global__ __launch_bounds__(WROWS* WCOLS * 32) void wmma_gemm_kernel(
    const void* __restrict__ Av, int lda, long long aSB, long long aSH,
    const unsigned short* __restrict__ Bv, int ldb, long long bSB, long long bSH,
    void* __restrict__ Cv, int ldc, long long cSB, long long cSH,
    int K, float scale)
{
    const int lane = threadIdx.x & 31;
    const int wave = threadIdx.x >> 5;
    const int zb   = blockIdx.z >> 4;
    const int zh   = blockIdx.z & 15;

    const int tileM0 = blockIdx.y * (WROWS * 64) + (wave / WCOLS) * 64;
    const int tileN0 = blockIdx.x * (WCOLS * 64) + (wave % WCOLS) * 64;

    const size_t aOff = (size_t)zb * (size_t)aSB + (size_t)zh * (size_t)aSH;
    const size_t bOff = (size_t)zb * (size_t)bSB + (size_t)zh * (size_t)bSH;
    const size_t cOff = (size_t)zb * (size_t)cSB + (size_t)zh * (size_t)cSH;

    const unsigned short* A16 = (const unsigned short*)Av + aOff;
    const float*          A32 = (const float*)Av + aOff;
    const unsigned short* B   = Bv + bOff;

    v8f acc[4][4];
#pragma unroll
    for (int mi = 0; mi < 4; ++mi)
#pragma unroll
        for (int ni = 0; ni < 4; ++ni)
            acc[mi][ni] = {};

    for (int k = 0; k < K; k += 32) {
        v16bf a[4], b[4];
#pragma unroll
        for (int mi = 0; mi < 4; ++mi) {
            if constexpr (A_F32)
                a[mi] = load_a_f32(A32, lda, tileM0 + mi * 16, k, lane);
            else
                a[mi] = load_a_bf16(A16, lda, tileM0 + mi * 16, k, lane);
        }
#pragma unroll
        for (int ni = 0; ni < 4; ++ni)
            b[ni] = load_b_bf16(B, ldb, tileN0 + ni * 16, k, lane);

#pragma unroll
        for (int mi = 0; mi < 4; ++mi)
#pragma unroll
            for (int ni = 0; ni < 4; ++ni)
                acc[mi][ni] = __builtin_amdgcn_wmma_f32_16x16x32_bf16(
                    false, a[mi], false, b[ni], (short)0, acc[mi][ni],
                    false, false);
    }

    // Epilogue. C f32 16x16 layout: VGPR r -> M = r (lanes 0-15) / r+8 (16-31),
    // N = lane & 15.
#pragma unroll
    for (int mi = 0; mi < 4; ++mi) {
#pragma unroll
        for (int ni = 0; ni < 4; ++ni) {
            const int m0 = tileM0 + mi * 16;
            const int n0 = tileN0 + ni * 16;
#pragma unroll
            for (int r = 0; r < 8; ++r) {
                const int   m = m0 + r + ((lane >> 4) << 3);
                const int   n = n0 + (lane & 15);
                const float v = acc[mi][ni][r] * scale;
                if constexpr (OUT_BF16) {
                    unsigned short* C = (unsigned short*)Cv + cOff;
                    if constexpr (TRANS_OUT)
                        C[(size_t)n * ldc + m] = f32_to_bf16_rne(v);
                    else
                        C[(size_t)m * ldc + n] = f32_to_bf16_rne(v);
                } else {
                    float* C = (float*)Cv + cOff;
                    C[(size_t)m * ldc + n] = v;
                }
            }
        }
    }
}

__global__ void convert_f32_bf16_kernel(const float* __restrict__ src,
                                        unsigned short* __restrict__ dst,
                                        long long n) {
    long long i = (long long)blockIdx.x * blockDim.x + threadIdx.x;
    if (i < n) dst[i] = f32_to_bf16_rne(src[i]);
}

// Row softmax in place over attn[row, 0..SEQ), with boolean mask.
// One block (256 threads) per row; 8 elements per thread kept in registers.
__global__ __launch_bounds__(256) void softmax_row_kernel(
    float* __restrict__ attn, const unsigned char* __restrict__ mask)
{
    const size_t row = blockIdx.x;                 // ((b*H + h) * SEQ + q)
    const int    q   = (int)(row & (SEQ - 1));
    const int    b   = (int)(row >> 15);           // row / (H*SEQ) = row / 32768
    float* r = attn + row * (size_t)SEQ;
    const unsigned char* mrow = mask + ((size_t)b * SEQ + q) * SEQ;

    const int tid = threadIdx.x;
    float v[8];
    float mx = -3.0e38f;
#pragma unroll
    for (int i = 0; i < 8; ++i) {
        const int c = tid + i * 256;
        float s = r[c];
        if (!mrow[c]) s = -1.0e9f;
        v[i] = s;
        mx = fmaxf(mx, s);
    }

    __shared__ float red[256];
    red[tid] = mx;
    __syncthreads();
    for (int off = 128; off > 0; off >>= 1) {
        if (tid < off) red[tid] = fmaxf(red[tid], red[tid + off]);
        __syncthreads();
    }
    mx = red[0];
    __syncthreads();

    float sum = 0.0f;
#pragma unroll
    for (int i = 0; i < 8; ++i) {
        v[i] = __expf(v[i] - mx);
        sum += v[i];
    }
    red[tid] = sum;
    __syncthreads();
    for (int off = 128; off > 0; off >>= 1) {
        if (tid < off) red[tid] += red[tid + off];
        __syncthreads();
    }
    const float inv = 1.0f / red[0];
#pragma unroll
    for (int i = 0; i < 8; ++i)
        r[tid + i * 256] = v[i] * inv;
}

extern "C" void kernel_launch(void* const* d_in, const int* in_sizes, int n_in,
                              void* d_out, int out_size, void* d_ws, size_t ws_size,
                              hipStream_t stream) {
    (void)in_sizes; (void)n_in; (void)out_size; (void)ws_size;

    const float*         x    = (const float*)d_in[0];
    const unsigned char* mask = (const unsigned char*)d_in[1];
    const float*         Wq   = (const float*)d_in[2];
    const float*         Wk   = (const float*)d_in[3];
    const float*         Wv   = (const float*)d_in[4];
    const float*         Wo   = (const float*)d_in[5];

    float* out  = (float*)d_out;                                   // [B,S,D]
    float* attn = out + (size_t)BATCH * SEQ * D_MODEL;             // [B,H,S,S]

    const size_t XN = (size_t)BATCH * SEQ * D_MODEL;   // 4 Mi elems
    const size_t WN = (size_t)D_MODEL * D_MODEL;       // 1 Mi elems
    unsigned short* xb  = (unsigned short*)d_ws;       // x bf16     [4096,1024]
    unsigned short* wqb = xb + XN;
    unsigned short* wkb = wqb + WN;
    unsigned short* wvb = wkb + WN;
    unsigned short* wob = wvb + WN;
    unsigned short* qb  = wob + WN;                    // Q bf16     [B,S,1024]
    unsigned short* kb  = qb + XN;                     // K bf16     [B,S,1024]
    unsigned short* vt  = kb + XN;                     // V^T bf16   [1024,4096]
    unsigned short* ctx = vt + XN;                     // ctx bf16   [B,S,1024]
    // total ws: 24 Mi ushorts = 48 MB

    // 1) fp32 -> bf16 conversions
    convert_f32_bf16_kernel<<<(unsigned)((XN + 255) / 256), 256, 0, stream>>>(x,  xb,  (long long)XN);
    convert_f32_bf16_kernel<<<(unsigned)((WN + 255) / 256), 256, 0, stream>>>(Wq, wqb, (long long)WN);
    convert_f32_bf16_kernel<<<(unsigned)((WN + 255) / 256), 256, 0, stream>>>(Wk, wkb, (long long)WN);
    convert_f32_bf16_kernel<<<(unsigned)((WN + 255) / 256), 256, 0, stream>>>(Wv, wvb, (long long)WN);
    convert_f32_bf16_kernel<<<(unsigned)((WN + 255) / 256), 256, 0, stream>>>(Wo, wob, (long long)WN);

    const dim3 blk128(128);
    const int  Mx = BATCH * SEQ;  // 4096

    // 2) Q = x@Wq^T, K = x@Wk^T (bf16 out);  V^T (bf16, transposed out, ldc=4096)
    wmma_gemm_kernel<2, 2, false, true, false>
        <<<dim3(D_MODEL / 128, Mx / 128, 1), blk128, 0, stream>>>(
        xb, D_MODEL, 0, 0, wqb, D_MODEL, 0, 0, qb, D_MODEL, 0, 0,
        D_MODEL, 1.0f);
    wmma_gemm_kernel<2, 2, false, true, false>
        <<<dim3(D_MODEL / 128, Mx / 128, 1), blk128, 0, stream>>>(
        xb, D_MODEL, 0, 0, wkb, D_MODEL, 0, 0, kb, D_MODEL, 0, 0,
        D_MODEL, 1.0f);
    wmma_gemm_kernel<2, 2, false, true, true>
        <<<dim3(D_MODEL / 128, Mx / 128, 1), blk128, 0, stream>>>(
        xb, D_MODEL, 0, 0, wvb, D_MODEL, 0, 0, vt, Mx, 0, 0,
        D_MODEL, 1.0f);

    // 3) scores = (1/sqrt(64)) * Q @ K^T, batched over 32 (b,h), f32 -> attn
    wmma_gemm_kernel<2, 2, false, false, false>
        <<<dim3(SEQ / 128, SEQ / 128, BATCH * N_HEADS), blk128, 0, stream>>>(
        qb, D_MODEL, (long long)SEQ * D_MODEL, HEAD_DIM,
        kb, D_MODEL, (long long)SEQ * D_MODEL, HEAD_DIM,
        attn, SEQ, (long long)N_HEADS * SEQ * SEQ, (long long)SEQ * SEQ,
        HEAD_DIM, 0.125f);

    // 4) softmax rows in place (mask applied)
    softmax_row_kernel<<<BATCH * N_HEADS * SEQ, 256, 0, stream>>>(attn, mask);

    // 5) ctx = attn @ V  (A read as f32, converted in-register; B = V^T rows)
    //    4x1 wave layout -> block tile 256x64; N = HEAD_DIM = 64 exactly.
    wmma_gemm_kernel<4, 1, true, true, false>
        <<<dim3(1, SEQ / 256, BATCH * N_HEADS), blk128, 0, stream>>>(
        attn, SEQ, (long long)N_HEADS * SEQ * SEQ, (long long)SEQ * SEQ,
        vt, Mx, (long long)SEQ, (long long)HEAD_DIM * Mx,
        ctx, D_MODEL, (long long)SEQ * D_MODEL, (long long)HEAD_DIM,
        SEQ, 1.0f);

    // 6) out = ctx @ Wo^T (f32 out)
    wmma_gemm_kernel<2, 2, false, false, false>
        <<<dim3(D_MODEL / 128, Mx / 128, 1), blk128, 0, stream>>>(
        ctx, D_MODEL, 0, 0, wob, D_MODEL, 0, 0, out, D_MODEL, 0, 0,
        D_MODEL, 1.0f);
}
